// GptOssGroupedExperts_54485955117200
// MI455X (gfx1250) — matrix operand order repaired and checked by reading, not compiled, not run on hardware
//
#include <hip/hip_runtime.h>
#include <hip/hip_bf16.h>

// ---------------------------------------------------------------------------
// GptOss grouped-experts MLP for gfx1250 (MI455X), bf16 WMMA path.
//   GEMM1: h = x_e * W1_e^T (+b1), fused swiglu, bf16 activation -> d_ws
//   GEMM2: out = act * W2_e^T (+b2), fp32 -> d_out
// Double-buffered LDS (1 barrier / K-step), v_cvt_pk_bf16_f32 staging,
// v_wmma_f32_16x16x32_bf16 compute.  GEMM2 stages its (already-bf16) A tile
// with GLOBAL_LOAD_ASYNC_TO_LDS_B128 (ASYNCcnt) instead of VGPR round-trips.
// ---------------------------------------------------------------------------

typedef __attribute__((ext_vector_type(16))) __bf16 v16bf;
typedef __attribute__((ext_vector_type(8)))  float  v8f;

#define ALPHA 1.702f
#define LIMIT 7.0f

#define LDT   34  // padded LDS row stride in halves (17 banks -> conflict free)
#define LDT_A 40  // GEMM2 A-tile stride: 80B rows, 16B aligned for async b128,
                  // 20-bank lane stride -> 16 distinct banks, conflict free

// scalar fp32 -> bf16 RNE (epilogue only)
__device__ __forceinline__ unsigned short f2bf(float f) {
    unsigned int u = __float_as_uint(f);
    u += 0x7FFFu + ((u >> 16) & 1u);
    return (unsigned short)(u >> 16);
}

// packed fp32x2 -> bf16x2 RNE in one VALU op (D[15:0]=cvt(S0), D[31:16]=cvt(S1))
__device__ __forceinline__ unsigned int cvt_pk_bf16(float lo, float hi) {
    unsigned int d;
    asm("v_cvt_pk_bf16_f32 %0, %1, %2" : "=v"(d) : "v"(lo), "v"(hi));
    return d;
}

// A fragment (16x32, M x K): lane%16 = M; laneHi + VGPR select K per ISA table.
__device__ __forceinline__ v16bf load_frag_a(const unsigned short* __restrict__ lds,
                                             int row_base, int lane, int ldt) {
    const int m  = lane & 15;
    const int hi = (lane >> 4) & 1;
    const unsigned short* p = lds + (row_base + m) * ldt;
    union { v16bf v; unsigned int u[8]; } f;
#pragma unroll
    for (int r = 0; r < 8; ++r) {
        const int k = ((r < 4) ? 0 : 16) + hi * 8 + 2 * (r & 3);
        f.u[r] = *(const unsigned int*)(p + k);
    }
    return f.v;
}

// B fragment (32x16, K x N) from an N-major LDS tile: lane%16 = N,
// laneHi + VGPR select K per ISA table (same as A).
__device__ __forceinline__ v16bf load_frag_b(const unsigned short* __restrict__ lds,
                                             int col_base, int lane) {
    const int n  = lane & 15;
    const int hi = (lane >> 4) & 1;
    const unsigned short* p = lds + (col_base + n) * LDT;
    union { v16bf v; unsigned int u[8]; } f;
#pragma unroll
    for (int r = 0; r < 8; ++r) {
        f.u[r] = *(const unsigned int*)(p + ((r < 4) ? 0 : 16) + hi * 8 + 2 * (r & 3));
    }
    return f.v;
}

#define WMMA_BF16(ACC, A, B) \
    ACC = __builtin_amdgcn_wmma_f32_16x16x32_bf16(false, A, false, B, (short)0, ACC, false, false)

// ---------------------------------------------------------------------------
// GEMM1 + swiglu.  Block tile 128 x 64(activated) (=128 h-cols), 8 waves 4x2,
// wave tile 32x32 with g+l accumulators -> 8 wmma/wave/K-step.
// grid = E * (C/128) * (H/64)
// ---------------------------------------------------------------------------
__global__ __launch_bounds__(256)
void gemm1_swiglu_kernel(const float* __restrict__ x,
                         const float* __restrict__ w1,
                         const float* __restrict__ b1,
                         unsigned short* __restrict__ hact,
                         int C, int D, int H, int n_mt, int n_nt) {
    __shared__ unsigned short sA [2][128 * LDT];
    __shared__ unsigned short sBg[2][ 64 * LDT];
    __shared__ unsigned short sBl[2][ 64 * LDT];

    const int bx = blockIdx.x;
    const int e  = bx / (n_mt * n_nt);
    const int r0 = bx % (n_mt * n_nt);
    const int m0 = (r0 / n_nt) * 128;
    const int n0 = (r0 % n_nt) * 64;
    const int twoH = 2 * H;

    const int t    = threadIdx.x;
    const int lane = t & 31;
    const int wid  = t >> 5;
    const int wm   = wid >> 1;        // 0..3  (M)
    const int wn   = wid & 1;         // 0..1  (N)
    const int r8   = t >> 3;          // 0..31 row within pass
    const int c4   = (t & 7) * 4;     // 0..28 col (floats)

    const float* xe = x  + (size_t)e * C * D;
    const float* we = w1 + (size_t)e * twoH * D;

    v8f acc_g[2][2], acc_l[2][2];
#pragma unroll
    for (int i = 0; i < 2; ++i)
#pragma unroll
        for (int j = 0; j < 2; ++j) { acc_g[i][j] = {}; acc_l[i][j] = {}; }

    float4 ar[4], bgr[2], blr[2];

    auto g_load = [&](int k0) {
#pragma unroll
        for (int pr = 0; pr < 4; ++pr)
            ar[pr] = *(const float4*)(xe + (size_t)(m0 + pr * 32 + r8) * D + k0 + c4);
#pragma unroll
        for (int pr = 0; pr < 2; ++pr) {
            const int n = n0 + pr * 32 + r8;
            bgr[pr] = *(const float4*)(we + (size_t)(2 * n)     * D + k0 + c4);
            blr[pr] = *(const float4*)(we + (size_t)(2 * n + 1) * D + k0 + c4);
        }
    };
    auto l_store = [&](int b) {
#pragma unroll
        for (int pr = 0; pr < 4; ++pr) {
            unsigned short* d = &sA[b][(pr * 32 + r8) * LDT + c4];
            *(unsigned int*)(d)     = cvt_pk_bf16(ar[pr].x, ar[pr].y);
            *(unsigned int*)(d + 2) = cvt_pk_bf16(ar[pr].z, ar[pr].w);
        }
#pragma unroll
        for (int pr = 0; pr < 2; ++pr) {
            unsigned short* dg = &sBg[b][(pr * 32 + r8) * LDT + c4];
            *(unsigned int*)(dg)     = cvt_pk_bf16(bgr[pr].x, bgr[pr].y);
            *(unsigned int*)(dg + 2) = cvt_pk_bf16(bgr[pr].z, bgr[pr].w);
            unsigned short* dl = &sBl[b][(pr * 32 + r8) * LDT + c4];
            *(unsigned int*)(dl)     = cvt_pk_bf16(blr[pr].x, blr[pr].y);
            *(unsigned int*)(dl + 2) = cvt_pk_bf16(blr[pr].z, blr[pr].w);
        }
    };

    const int nK = D / 32;
    g_load(0);
    l_store(0);

    for (int kt = 0; kt < nK; ++kt) {
        __syncthreads();
        const bool more = (kt + 1) < nK;
        if (more) g_load((kt + 1) * 32);

        const int buf = kt & 1;
        v16bf a0  = load_frag_a(sA[buf],  wm * 32,      lane, LDT);
        v16bf a1  = load_frag_a(sA[buf],  wm * 32 + 16, lane, LDT);
        v16bf bg0 = load_frag_b(sBg[buf], wn * 32,      lane);
        v16bf bg1 = load_frag_b(sBg[buf], wn * 32 + 16, lane);
        v16bf bl0 = load_frag_b(sBl[buf], wn * 32,      lane);
        v16bf bl1 = load_frag_b(sBl[buf], wn * 32 + 16, lane);

        WMMA_BF16(acc_g[0][0], a0, bg0);
        WMMA_BF16(acc_g[0][1], a0, bg1);
        WMMA_BF16(acc_g[1][0], a1, bg0);
        WMMA_BF16(acc_g[1][1], a1, bg1);
        WMMA_BF16(acc_l[0][0], a0, bl0);
        WMMA_BF16(acc_l[0][1], a0, bl1);
        WMMA_BF16(acc_l[1][0], a1, bl0);
        WMMA_BF16(acc_l[1][1], a1, bl1);

        if (more) l_store((kt + 1) & 1);
    }

    // ---- epilogue: bias + swiglu, store bf16 activation ----
    const int hi = (lane >> 4) & 1;
    const int ln = lane & 15;
#pragma unroll
    for (int tm = 0; tm < 2; ++tm) {
#pragma unroll
        for (int tn = 0; tn < 2; ++tn) {
            const int gn = n0 + wn * 32 + tn * 16 + ln;        // activated column
            const float bg = b1[(size_t)e * twoH + 2 * gn];
            const float bl = b1[(size_t)e * twoH + 2 * gn + 1];
            size_t idx = ((size_t)(e * C + m0 + wm * 32 + tm * 16 + 8 * hi)) * H + gn;
#pragma unroll
            for (int v = 0; v < 8; ++v) {
                float g = acc_g[tm][tn][v] + bg;
                float l = acc_l[tm][tn][v] + bl;
                g = fminf(g, LIMIT);
                l = fminf(fmaxf(l, -LIMIT), LIMIT);
                const float sig = 1.0f / (1.0f + __expf(-ALPHA * g));
                hact[idx] = f2bf(g * sig * (l + 1.0f));
                idx += (size_t)H;
            }
        }
    }
}

// ---------------------------------------------------------------------------
// GEMM2.  Block tile 128x96, 8 waves 4x2, wave tile 32x48 (2x3 wmma tiles)
// -> 6 wmma/wave/K-step.  A tile staged via async global->LDS DMA.
// grid = E * (C/128) * (D/96)
// ---------------------------------------------------------------------------
#define SA_HALVES (128 * LDT_A)            // one A buffer, in halves
#define SB_HALVES ( 96 * LDT)              // one B buffer, in halves

__global__ __launch_bounds__(256)
void gemm2_kernel(const unsigned short* __restrict__ hact,
                  const float* __restrict__ w2,
                  const float* __restrict__ b2,
                  float* __restrict__ out,
                  int C, int D, int H, int n_mt, int n_nt) {
    // Single LDS allocation so the async-DMA raw offsets start at 0.
    __shared__ unsigned short smem[2 * SA_HALVES + 2 * SB_HALVES];

    const int bx = blockIdx.x;
    const int e  = bx / (n_mt * n_nt);
    const int r0 = bx % (n_mt * n_nt);
    const int m0 = (r0 / n_nt) * 128;
    const int n0 = (r0 % n_nt) * 96;

    const int t    = threadIdx.x;
    const int lane = t & 31;
    const int wid  = t >> 5;
    const int wm   = wid >> 1;        // 0..3 (M)
    const int wn   = wid & 1;         // 0..1 (N, 48-wide)
    const int r8   = t >> 3;
    const int c4   = (t & 7) * 4;

    const unsigned short* ae = hact + (size_t)e * C * H;
    const float*          we = w2   + (size_t)e * D * H;

    v8f acc[2][3];
#pragma unroll
    for (int i = 0; i < 2; ++i)
#pragma unroll
        for (int j = 0; j < 3; ++j) acc[i][j] = {};

    float4 br[3];

    // async DMA of the 128x32 bf16 A tile: 2 x b128 per thread
    auto issue_async_a = [&](int b, int k0) {
#pragma unroll
        for (int pr = 0; pr < 2; ++pr) {
            const int row   = pr * 64 + (t >> 2);
            const int col   = (t & 3) * 8;                       // halves (16B)
            const unsigned long long g =
                (unsigned long long)(ae + (size_t)(m0 + row) * H + k0 + col);
            const unsigned int lds_off =
                (unsigned int)(b * SA_HALVES + row * LDT_A + col) * 2u;  // bytes
            asm volatile("global_load_async_to_lds_b128 %0, %1, off"
                         :: "v"(lds_off), "v"(g) : "memory");
        }
    };
    auto g_load_b = [&](int k0) {
#pragma unroll
        for (int pr = 0; pr < 3; ++pr)
            br[pr] = *(const float4*)(we + (size_t)(n0 + pr * 32 + r8) * H + k0 + c4);
    };
    auto l_store_b = [&](int b) {
        unsigned short* sB = smem + 2 * SA_HALVES + b * SB_HALVES;
#pragma unroll
        for (int pr = 0; pr < 3; ++pr) {
            unsigned short* d = sB + (pr * 32 + r8) * LDT + c4;
            *(unsigned int*)(d)     = cvt_pk_bf16(br[pr].x, br[pr].y);
            *(unsigned int*)(d + 2) = cvt_pk_bf16(br[pr].z, br[pr].w);
        }
    };

    const int nK = H / 32;
    issue_async_a(0, 0);
    g_load_b(0);
    l_store_b(0);

    for (int kt = 0; kt < nK; ++kt) {
        asm volatile("s_wait_asynccnt 0" ::: "memory");   // A tile for `kt` landed
        __syncthreads();
        const bool more = (kt + 1) < nK;
        if (more) {
            issue_async_a((kt + 1) & 1, (kt + 1) * 32);
            g_load_b((kt + 1) * 32);
        }

        const int buf = kt & 1;
        const unsigned short* sA = smem + buf * SA_HALVES;
        const unsigned short* sB = smem + 2 * SA_HALVES + buf * SB_HALVES;
        v16bf a0  = load_frag_a(sA, wm * 32,      lane, LDT_A);
        v16bf a1  = load_frag_a(sA, wm * 32 + 16, lane, LDT_A);
        v16bf b0  = load_frag_b(sB, wn * 48,      lane);
        v16bf b1f = load_frag_b(sB, wn * 48 + 16, lane);
        v16bf b2f = load_frag_b(sB, wn * 48 + 32, lane);

        WMMA_BF16(acc[0][0], a0, b0);
        WMMA_BF16(acc[0][1], a0, b1f);
        WMMA_BF16(acc[0][2], a0, b2f);
        WMMA_BF16(acc[1][0], a1, b0);
        WMMA_BF16(acc[1][1], a1, b1f);
        WMMA_BF16(acc[1][2], a1, b2f);

        if (more) l_store_b((kt + 1) & 1);
    }

    // ---- epilogue: bias2, fp32 store ----
    const int hi = (lane >> 4) & 1;
    const int ln = lane & 15;
#pragma unroll
    for (int tm = 0; tm < 2; ++tm) {
#pragma unroll
        for (int tn = 0; tn < 3; ++tn) {
            const int gn = n0 + wn * 48 + tn * 16 + ln;
            const float bias = b2[(size_t)e * D + gn];
            size_t idx = ((size_t)(e * C + m0 + wm * 32 + tm * 16 + 8 * hi)) * D + gn;
#pragma unroll
            for (int v = 0; v < 8; ++v) {
                out[idx] = acc[tm][tn][v] + bias;
                idx += (size_t)D;
            }
        }
    }
}

// ---------------------------------------------------------------------------
extern "C" void kernel_launch(void* const* d_in, const int* in_sizes, int n_in,
                              void* d_out, int out_size, void* d_ws, size_t ws_size,
                              hipStream_t stream) {
    const float* x  = (const float*)d_in[0];
    // d_in[1] = num_tokens_per_expert (uniform T/E split; reference reshapes uniformly)
    const float* w1 = (const float*)d_in[2];
    const float* b1 = (const float*)d_in[3];
    const float* w2 = (const float*)d_in[4];
    const float* b2 = (const float*)d_in[5];
    float* out = (float*)d_out;

    const int E    = in_sizes[1];                 // 8
    const int twoH = in_sizes[3] / E;             // 5760
    const int H    = twoH / 2;                    // 2880
    const int D    = in_sizes[2] / in_sizes[3];   // 2880
    const int T    = in_sizes[0] / D;             // 16384
    const int C    = T / E;                       // 2048

    unsigned short* hact = (unsigned short*)d_ws; // T*H bf16 activations

    const int mt  = C / 128;
    const int nt1 = H / 64;
    const int nt2 = D / 96;

    gemm1_swiglu_kernel<<<E * mt * nt1, 256, 0, stream>>>(x, w1, b1, hact, C, D, H, mt, nt1);
    gemm2_kernel<<<E * mt * nt2, 256, 0, stream>>>(hact, w2, b2, out, C, D, H, mt, nt2);
}
